// GNN_decoder_49400713838638
// MI455X (gfx1250) — matrix operand order repaired and checked by reading, not compiled, not run on hardware
//
#include <hip/hip_runtime.h>
#include <hip/hip_bf16.h>

typedef __attribute__((ext_vector_type(16))) _Float16 v16h;
typedef __attribute__((ext_vector_type(8)))  float    v8f;

#define NODES 10000
#define EDGES 320000

// ---------------------------------------------------------------------------
// f32 -> f16 conversion (vectorized x4). n % 4 == 0 for all calls here.
// ---------------------------------------------------------------------------
__global__ __launch_bounds__(256)
void cvt_f16_kernel(const float* __restrict__ in, _Float16* __restrict__ out,
                    long long n4)
{
    const long long i = (long long)blockIdx.x * blockDim.x + threadIdx.x;
    if (i >= n4) return;
    float4 v = ((const float4*)in)[i];
    _Float16 h[4] = { (_Float16)v.x, (_Float16)v.y, (_Float16)v.z, (_Float16)v.w };
    ((ulong1*)out)[i] = *(ulong1*)h;   // 8-byte store of 4 halves
}

// ---------------------------------------------------------------------------
// GEMM: OUT[M x D] = epilogue( X[M x K] @ W[D x K]^T ), f16 in / f32 acc.
// One wave32 owns a 64x128 output band: 4 (m) x 8 (d) subtiles of 16x16,
// 32 accumulators, 32 v_wmma_f32_16x16x32_f16 per K-step of 32.
// Arithmetic intensity: 524 KFLOP / 12 KB loaded = ~43 FLOP/B from L2.
// Band origins clamped to M-64 / D-128 for ragged edges (overlap recomputes
// identical values -> benign). Requires M >= 64, D >= 128, K % 32 == 0.
// ---------------------------------------------------------------------------
#define MI 4
#define NJ 8

__global__ __launch_bounds__(256)
void gemm_wmma_kernel(const _Float16* __restrict__ X,
                      const _Float16* __restrict__ W,
                      float* __restrict__ OUT,
                      int M, int K, int D,
                      const int* __restrict__ active,
                      int do_leaky, int do_sigmoid)
{
    const int lane = threadIdx.x & 31;
    const int wave = threadIdx.x >> 5;
    const int waves_per_blk = blockDim.x >> 5;

    const int bands_m = (M + 63) >> 6;
    const int bands_d = (D + 127) >> 7;
    const long long band = (long long)blockIdx.x * waves_per_blk + wave;
    if (band >= (long long)bands_m * bands_d) return;   // wave-uniform exit

    const int bm = (int)(band / bands_d);
    const int bd = (int)(band % bands_d);
    const int m0 = min(bm * 64,  M - 64);
    const int d0 = min(bd * 128, D - 128);

    const int r  = lane & 15;     // row (A) / column (B) within a 16-subtile
    const int hi = lane >> 4;     // which K-half this lane owns

    const _Float16* aptr[MI];
    const _Float16* bptr[NJ];
#pragma unroll
    for (int i = 0; i < MI; ++i)
        aptr[i] = X + (size_t)(m0 + i * 16 + r) * K;
#pragma unroll
    for (int j = 0; j < NJ; ++j)
        bptr[j] = W + (size_t)(d0 + j * 16 + r) * K;

    v8f c[MI][NJ] = {};

    for (int k0 = 0; k0 < K; k0 += 32) {
        union Frag { v16h v; float4 q[2]; };
        Frag A[MI], B[NJ];
#pragma unroll
        for (int i = 0; i < MI; ++i) {
            // A layout (16-bit, 16x32): lane<16 -> K {k0..+7, k0+16..+23};
            //                           lane>=16 -> K {k0+8..+15, k0+24..+31}
            A[i].q[0] = *(const float4*)(aptr[i] + k0 + hi * 8);
            A[i].q[1] = *(const float4*)(aptr[i] + k0 + 16 + hi * 8);
        }
#pragma unroll
        for (int j = 0; j < NJ; ++j) {
            // B layout (32x16): lane L holds col d0+16j+(L%16), K {k0+hi*8..+7, k0+16+hi*8..+7}
            B[j].q[0] = *(const float4*)(bptr[j] + k0 + hi * 8);
            B[j].q[1] = *(const float4*)(bptr[j] + k0 + 16 + hi * 8);
        }
#pragma unroll
        for (int i = 0; i < MI; ++i)
#pragma unroll
            for (int j = 0; j < NJ; ++j)
                c[i][j] = __builtin_amdgcn_wmma_f32_16x16x32_f16(
                    false, A[i].v, false, B[j].v,
                    (short)0, c[i][j], false, false);
    }

    // Epilogue. C layout: element ii of lane L -> row +hi*8+ii, col +(L%16).
    const bool lk = do_leaky && (*active != 0);
#pragma unroll
    for (int i = 0; i < MI; ++i) {
#pragma unroll
        for (int j = 0; j < NJ; ++j) {
#pragma unroll
            for (int ii = 0; ii < 8; ++ii) {
                float v = c[i][j][ii];
                if (lk) v = (v >= 0.0f) ? v : 0.01f * v;
                if (do_sigmoid) v = __builtin_amdgcn_rcpf(1.0f + __expf(-v));
                OUT[(size_t)(m0 + i * 16 + hi * 8 + ii) * D + (d0 + j * 16 + r)] = v;
            }
        }
    }
}

// ---------------------------------------------------------------------------
// SPMM: out[row[e], :] += vals[e] * sup[col[e], :]   (out pre-zeroed)
// Thread per (edge, feature); global_atomic_add_f32; traffic is L2-resident
// (support matrix <= 20 MB << 192 MB L2).
// ---------------------------------------------------------------------------
__global__ __launch_bounds__(256)
void spmm_kernel(const int* __restrict__ row, const int* __restrict__ col,
                 const float* __restrict__ vals, const float* __restrict__ sup,
                 float* __restrict__ out, int E, int D)
{
    const long long tid = (long long)blockIdx.x * blockDim.x + threadIdx.x;
    const long long total = (long long)E * D;
    if (tid >= total) return;
    const int e = (int)(tid / D);
    const int d = (int)(tid % D);
    const int rr = row[e];
    const int cc = col[e];
    const float m = vals[e] * sup[(size_t)cc * D + d];
    __hip_atomic_fetch_add(out + (size_t)rr * D + d, m,
                           __ATOMIC_RELAXED, __HIP_MEMORY_SCOPE_AGENT);
}

// ---------------------------------------------------------------------------
extern "C" void kernel_launch(void* const* d_in, const int* in_sizes, int n_in,
                              void* d_out, int out_size, void* d_ws, size_t ws_size,
                              hipStream_t stream)
{
    const float* z      = (const float*)d_in[0];
    const int*   arow   = (const int*)  d_in[1];
    const int*   acol   = (const int*)  d_in[2];
    const float* avals  = (const float*)d_in[3];
    const float* w1     = (const float*)d_in[4];
    const float* w2     = (const float*)d_in[5];
    const float* w3     = (const float*)d_in[6];
    const int*   active = (const int*)  d_in[7];

    const int N = NODES, E = EDGES;

    float* xhat = (float*)d_out;                    // [N, 512]
    float* adj  = (float*)d_out + (size_t)N * 512;  // [N, N]

    // Workspace carve-up (all chunks 16B aligned).
    char* p = (char*)d_ws;
    float* sup = (float*)p;          p += (size_t)N * 512 * sizeof(float);
    float* x1  = (float*)p;          p += (size_t)N * 128 * sizeof(float);
    float* x2  = (float*)p;          p += (size_t)N * 256 * sizeof(float);
    _Float16* z16    = (_Float16*)p; p += (size_t)N * 64  * sizeof(_Float16);
    _Float16* x1_16  = (_Float16*)p; p += (size_t)N * 128 * sizeof(_Float16);
    _Float16* x2_16  = (_Float16*)p; p += (size_t)N * 256 * sizeof(_Float16);
    _Float16* xh16   = (_Float16*)p; p += (size_t)N * 512 * sizeof(_Float16);
    _Float16* w1_16  = (_Float16*)p; p += (size_t)128 * 64  * sizeof(_Float16);
    _Float16* w2_16  = (_Float16*)p; p += (size_t)256 * 128 * sizeof(_Float16);
    _Float16* w3_16  = (_Float16*)p; p += (size_t)512 * 256 * sizeof(_Float16);

    auto cvt = [&](const float* in, _Float16* out, long long n) {
        long long n4 = n / 4;
        hipLaunchKernelGGL(cvt_f16_kernel, dim3((unsigned)((n4 + 255) / 256)),
                           dim3(256), 0, stream, in, out, n4);
    };
    auto gemm = [&](const _Float16* X, const _Float16* Wm, float* O,
                    int K, int D, int leaky, int sig) {
        long long bands  = (long long)((N + 63) / 64) * ((D + 127) / 128);
        long long blocks = (bands + 7) / 8;         // 8 waves / 256-thread block
        hipLaunchKernelGGL(gemm_wmma_kernel, dim3((unsigned)blocks), dim3(256),
                           0, stream, X, Wm, O, N, K, D, active, leaky, sig);
    };
    auto spmm = [&](const float* S, float* O, int D) {
        hipMemsetAsync(O, 0, (size_t)N * D * sizeof(float), stream);
        long long tot = (long long)E * D;
        hipLaunchKernelGGL(spmm_kernel, dim3((unsigned)((tot + 255) / 256)),
                           dim3(256), 0, stream, arow, acol, avals, S, O, E, D);
    };

    // One-time operand conversions (tiny).
    cvt(z,  z16,   (long long)N * 64);
    cvt(w1, w1_16, 128LL * 64);
    cvt(w2, w2_16, 256LL * 128);
    cvt(w3, w3_16, 512LL * 256);

    // Layer 1: support = lrelu(z @ w1^T) [N,128]; x1 = A @ support
    gemm(z16, w1_16, sup, 64, 128, 1, 0);
    spmm(sup, x1, 128);
    cvt(x1, x1_16, (long long)N * 128);
    // Layer 2
    gemm(x1_16, w2_16, sup, 128, 256, 1, 0);
    spmm(sup, x2, 256);
    cvt(x2, x2_16, (long long)N * 256);
    // Layer 3 -> x_hat
    gemm(x2_16, w3_16, sup, 256, 512, 1, 0);
    spmm(sup, xhat, 512);
    cvt(xhat, xh16, (long long)N * 512);
    // adj = sigmoid(x_hat @ x_hat^T)
    gemm(xh16, xh16, adj, 512, N, 0, 1);
}
